// Gemma4DecoderLayer_75849122447855
// MI455X (gfx1250) — matrix operand order repaired and checked by reading, not compiled, not run on hardware
//
#include <hip/hip_runtime.h>
#include <stdint.h>

// ---------------- problem constants ----------------
#define BB   4
#define SS   1024
#define DD   2048
#define HQ_  16
#define HKV_ 8
#define HD_  128
#define GRP_ 2
#define EE   8
#define FE_  1024
#define FS_  8192
#define M4   4096        // B*S
#define EPSF 1e-6f

// ---------------- vector types ----------------
typedef __attribute__((ext_vector_type(16))) __bf16 v16bf;
typedef __attribute__((ext_vector_type(8)))  __bf16 v8bf;
typedef __attribute__((ext_vector_type(8)))  float  v8f;

__device__ __forceinline__ v16bf cat16(v8bf lo, v8bf hi) {
  return __builtin_shufflevector(lo, hi, 0,1,2,3,4,5,6,7,8,9,10,11,12,13,14,15);
}
__device__ __forceinline__ v8f wmma_bf16(v16bf a, v16bf b, v8f c) {
  return __builtin_amdgcn_wmma_f32_16x16x32_bf16(false, a, false, b, (short)0, c, false, false);
}
__device__ __forceinline__ v8f zero8() {
  v8f z = {0.f,0.f,0.f,0.f,0.f,0.f,0.f,0.f};
  return z;
}
__device__ __forceinline__ unsigned lds_addr_of(const void* p) {
  // generic LDS pointer: low 32 bits are the LDS byte offset (aperture rule)
  return (unsigned)(unsigned long long)p;
}

// ---------------- elementwise helpers ----------------
__global__ void cvt_kernel(const float* __restrict__ in, __bf16* __restrict__ out, long long n) {
  long long i = (long long)blockIdx.x * blockDim.x + threadIdx.x;
  long long stride = (long long)gridDim.x * blockDim.x;
  for (; i < n; i += stride) out[i] = (__bf16)in[i];
}

__global__ void zero_kernel(float* __restrict__ p, long long n) {
  long long i = (long long)blockIdx.x * blockDim.x + threadIdx.x;
  long long stride = (long long)gridDim.x * blockDim.x;
  for (; i < n; i += stride) p[i] = 0.0f;
}

__device__ __forceinline__ float gelu_tanh(float x) {
  float x3 = x * x * x;
  return 0.5f * x * (1.0f + tanhf(0.7978845608028654f * (x + 0.044715f * x3)));
}

__global__ void geglu_kernel(const float* __restrict__ g0, const float* __restrict__ g1,
                             __bf16* __restrict__ out, long long n) {
  long long i = (long long)blockIdx.x * blockDim.x + threadIdx.x;
  long long stride = (long long)gridDim.x * blockDim.x;
  for (; i < n; i += stride) out[i] = (__bf16)(gelu_tanh(g0[i]) * g1[i]);
}

// rms over row of `cols`; optional scale, residual add; writes f32 and/or bf16
__global__ void rmsnorm_kernel(const float* __restrict__ in, const float* __restrict__ scale,
                               const float* __restrict__ addend,
                               float* __restrict__ out_f32, __bf16* __restrict__ out_bf16,
                               int cols, float mul) {
  int row = blockIdx.x;
  const float* p = in + (size_t)row * cols;
  int tid = threadIdx.x;
  float ss = 0.0f;
  for (int c = tid; c < cols; c += 256) { float v = p[c]; ss += v * v; }
  for (int m = 1; m < 32; m <<= 1) ss += __shfl_xor(ss, m, 32);
  __shared__ float sred[8];
  if ((tid & 31) == 0) sred[tid >> 5] = ss;
  __syncthreads();
  float tot = 0.0f;
#pragma unroll
  for (int i = 0; i < 8; i++) tot += sred[i];
  float r = rsqrtf(tot / (float)cols + EPSF) * mul;
  for (int c = tid; c < cols; c += 256) {
    float y = p[c] * r;
    if (scale)  y *= scale[c];
    if (addend) y += addend[(size_t)row * cols + c];
    if (out_f32)  out_f32[(size_t)row * cols + c] = y;
    if (out_bf16) out_bf16[(size_t)row * cols + c] = (__bf16)y;
  }
}

// per-head RMS norm (+ optional RoPE), 128 threads = HD
__global__ void headnorm_rope_kernel(const float* __restrict__ in, const float* __restrict__ scale,
                                     const int* __restrict__ positions,
                                     __bf16* __restrict__ out, int H, int do_rope) {
  int blk = blockIdx.x;
  int row = blk / H, hh = blk % H;           // row = b*S + s
  const float* p = in + ((size_t)row * H + hh) * HD_;
  __bf16* o = out + ((size_t)row * H + hh) * HD_;
  int tid = threadIdx.x;                     // 0..127
  float v = p[tid];
  float ss = v * v;
  for (int m = 1; m < 32; m <<= 1) ss += __shfl_xor(ss, m, 32);
  __shared__ float sred[4];
  __shared__ float sy[128];
  if ((tid & 31) == 0) sred[tid >> 5] = ss;
  __syncthreads();
  float tot = sred[0] + sred[1] + sred[2] + sred[3];
  float y = v * rsqrtf(tot * (1.0f / 128.0f) + EPSF) * scale[tid];
  if (!do_rope) { o[tid] = (__bf16)y; return; }
  sy[tid] = y;
  __syncthreads();
  if (tid < 64) {
    float pos = (float)positions[row];
    float inv = __expf(-(float)tid * (9.210340371976184f / 64.0f)); // 10000^(-i/64)
    float ang = pos * inv;
    float s, c;
    __sincosf(ang, &s, &c);
    float x1 = sy[tid], x2 = sy[tid + 64];
    o[tid]      = (__bf16)(x1 * c - x2 * s);
    o[tid + 64] = (__bf16)(x2 * c + x1 * s);
  }
}

// ---------------- WMMA GEMM: C[f32] = A[bf16, MxK] @ B[bf16, KxN] ----------------
// 128x128 tile per 256-thread block (8 waves x 16-row strips), K-step 32.
// Double-buffered LDS; A tile staged with GLOBAL_LOAD_ASYNC_TO_LDS_B128 (ASYNCcnt);
// B tile transposed in-flight (manual); next-next tile prefetched.
__global__ __launch_bounds__(256) void gemm_bf16_kernel(
    const __bf16* __restrict__ A, const __bf16* __restrict__ Bm,
    float* __restrict__ C, int M, int N, int K) {
  __shared__ __align__(16) __bf16 As[2][128 * 32];   // [m][k]
  __shared__ __align__(16) __bf16 Bt[2][128 * 32];   // [n][k] (transposed)
  int tid = threadIdx.x;
  int lane = tid & 31, wid = tid >> 5;
  int h = (lane >> 4) & 1, lm = lane & 15;
  int m0 = blockIdx.y * 128, n0 = blockIdx.x * 128;

  v8f acc[8];
#pragma unroll
  for (int i = 0; i < 8; i++) acc[i] = zero8();

  int arow = tid >> 1, ac0 = (tid & 1) * 16;  // A staging: 128 rows x 32 cols
  int bk = tid >> 3, bn0 = (tid & 7) * 16;    // B staging: 32 rows x 128 cols

  const __bf16* gaBase = A + (size_t)(m0 + arow) * K + ac0;
  const __bf16* gbBase = Bm + (size_t)bk * N + n0 + bn0;

  auto stage = [&](int k0, int buf) {
    // A tile: async global->LDS copy, 32 bytes per thread (2 x B128)
    unsigned l0 = lds_addr_of(&As[buf][arow * 32 + ac0]);
    const __bf16* ga = gaBase + k0;
    asm volatile(
        "global_load_async_to_lds_b128 %0, %2, off\n\t"
        "global_load_async_to_lds_b128 %1, %3, off"
        :: "v"(l0), "v"(l0 + 16u), "v"(ga), "v"(ga + 8)
        : "memory");
    // B tile: load row-major, store transposed
    const __bf16* gb = gbBase + (size_t)k0 * N;
    v8bf lo = *(const v8bf*)gb;
    v8bf hi = *(const v8bf*)(gb + 8);
#pragma unroll
    for (int i = 0; i < 8; i++) Bt[buf][(bn0 + i) * 32 + bk]     = lo[i];
#pragma unroll
    for (int i = 0; i < 8; i++) Bt[buf][(bn0 + 8 + i) * 32 + bk] = hi[i];
  };

  stage(0, 0);
  asm volatile("s_wait_asynccnt 0x0" ::: "memory");
  __syncthreads();

  int buf = 0;
  for (int k0 = 0; k0 < K; k0 += 32) {
    int kn = k0 + 32;
    if (kn < K) {
      stage(kn, buf ^ 1);
      if (kn + 32 < K) {  // prefetch tile after next into L2
        __builtin_prefetch(gaBase + kn + 32, 0, 1);
        __builtin_prefetch(gbBase + (size_t)(kn + 32) * N, 0, 1);
      }
    }

    // A fragment for this wave's 16-row strip (CDNA5 16-bit A layout)
    const __bf16* ar = &As[buf][(wid * 16 + lm) * 32];
    v8bf alo = *(const v8bf*)(ar + 8 * h);
    v8bf ahi = *(const v8bf*)(ar + 8 * h + 16);
    v16bf afrag = cat16(alo, ahi);

    // batch all B-fragment loads, then issue WMMAs back-to-back
    v16bf bfr[8];
#pragma unroll
    for (int ct = 0; ct < 8; ct++) {
      const __bf16* br = &Bt[buf][(ct * 16 + lm) * 32 + 8 * h];
      v8bf blo = *(const v8bf*)br;
      v8bf bhi = *(const v8bf*)(br + 16);
      bfr[ct] = cat16(blo, bhi);
    }
#pragma unroll
    for (int ct = 0; ct < 8; ct++) acc[ct] = wmma_bf16(afrag, bfr[ct], acc[ct]);

    asm volatile("s_wait_asynccnt 0x0" ::: "memory");
    __syncthreads();
    buf ^= 1;
  }

  // epilogue: C layout — vgpr r, lanes 0-15: M=r ; lanes 16-31: M=r+8 ; N = lane%16
#pragma unroll
  for (int ct = 0; ct < 8; ct++) {
#pragma unroll
    for (int r = 0; r < 8; r++) {
      int row = m0 + wid * 16 + r + 8 * h;
      int col = n0 + ct * 16 + lm;
      C[(size_t)row * N + col] = acc[ct][r];
    }
  }
}

// ---------------- flash attention (causal, GQA), WMMA bf16 ----------------
// grid: (S/128, B*HQ); 256 threads = 8 waves, each wave owns 16 q rows.
// KV tiles of 64. Q fragments in registers; K read as B-fragments straight
// from global; V staged transposed in LDS; P staged per-wave in LDS.
__global__ __launch_bounds__(256) void attn_kernel(
    const __bf16* __restrict__ Q,   // (B,S,HQ,HD)
    const __bf16* __restrict__ Kk,  // (B,S,HKV,HD)
    const __bf16* __restrict__ V,   // (B,S,HKV,HD)
    __bf16* __restrict__ O) {       // (B,S,HQ,HD)
  __shared__ __align__(16) __bf16 Ps[128 * 64];   // [qrow_local][kv_local]
  __shared__ __align__(16) __bf16 Vt[128 * 64];   // [hd][kv_local]
  int tid = threadIdx.x, lane = tid & 31, wid = tid >> 5;
  int h = lane >> 4, lm = lane & 15;
  int bh = blockIdx.y;
  int b = bh / HQ_, hq = bh % HQ_;
  int kh = hq / GRP_;
  int q0 = blockIdx.x * 128;

  // Q fragments (4 k-steps over HD=128)
  int qrow = q0 + wid * 16 + lm;
  const __bf16* qp = Q + ((size_t)(b * SS + qrow) * HQ_ + hq) * HD_;
  v16bf qf[4];
#pragma unroll
  for (int ks = 0; ks < 4; ks++) {
    v8bf lo = *(const v8bf*)(qp + ks * 32 + 8 * h);
    v8bf hi = *(const v8bf*)(qp + ks * 32 + 8 * h + 16);
    qf[ks] = cat16(lo, hi);
  }

  v8f oacc[8];
#pragma unroll
  for (int i = 0; i < 8; i++) oacc[i] = zero8();
  float mrow[8], lrow[8];
#pragma unroll
  for (int r = 0; r < 8; r++) { mrow[r] = -3.0e38f; lrow[r] = 0.0f; }

  int ntile = q0 / 64 + 2;   // covers kv up to q0+127
  for (int t = 0; t < ntile; t++) {
    int kv0 = t * 64;
    __syncthreads();
    // stage V transposed: Vt[hd][kv]
    {
      int kr = tid >> 2;             // kv row 0..63
      int hd0 = (tid & 3) * 32;      // 0,32,64,96
      const __bf16* vp = V + ((size_t)(b * SS + kv0 + kr) * HKV_ + kh) * HD_ + hd0;
#pragma unroll
      for (int j = 0; j < 4; j++) {
        v8bf vv = *(const v8bf*)(vp + j * 8);
#pragma unroll
        for (int i = 0; i < 8; i++) Vt[(hd0 + j * 8 + i) * 64 + kr] = vv[i];
      }
      if (t + 1 < ntile) {  // prefetch next KV tile's V rows
        __builtin_prefetch(vp + (size_t)64 * HKV_ * HD_, 0, 1);
      }
    }
    __syncthreads();

    // S = Q @ K^T for this wave: 4 column tiles of 16
    v8f sacc[4];
#pragma unroll
    for (int i = 0; i < 4; i++) sacc[i] = zero8();
#pragma unroll
    for (int ct = 0; ct < 4; ct++) {
      int kvcol = kv0 + ct * 16 + lm;          // this lane's B column (kv position)
      const __bf16* kp = Kk + ((size_t)(b * SS + kvcol) * HKV_ + kh) * HD_;
      v16bf kfr[4];
#pragma unroll
      for (int ks = 0; ks < 4; ks++) {
        v8bf lo = *(const v8bf*)(kp + ks * 32 + 8 * h);
        v8bf hi = *(const v8bf*)(kp + ks * 32 + 8 * h + 16);
        kfr[ks] = cat16(lo, hi);
      }
#pragma unroll
      for (int ks = 0; ks < 4; ks++) sacc[ct] = wmma_bf16(qf[ks], kfr[ks], sacc[ct]);
    }

    // causal mask + tile row max
    float tmax[8];
#pragma unroll
    for (int r = 0; r < 8; r++) tmax[r] = -3.0e38f;
#pragma unroll
    for (int ct = 0; ct < 4; ct++) {
#pragma unroll
      for (int r = 0; r < 8; r++) {
        int qr = q0 + wid * 16 + r + 8 * h;
        int kc = kv0 + ct * 16 + lm;
        float s = sacc[ct][r];
        if (kc > qr) { s = -3.0e38f; sacc[ct][r] = s; }
        tmax[r] = fmaxf(tmax[r], s);
      }
    }
#pragma unroll
    for (int r = 0; r < 8; r++)
      for (int m = 1; m < 16; m <<= 1) tmax[r] = fmaxf(tmax[r], __shfl_xor(tmax[r], m, 32));

    // online softmax rescale
    float alpha[8];
#pragma unroll
    for (int r = 0; r < 8; r++) {
      float mn = fmaxf(mrow[r], tmax[r]);
      alpha[r] = __expf(mrow[r] - mn);
      mrow[r] = mn;
      lrow[r] *= alpha[r];
    }
#pragma unroll
    for (int nt = 0; nt < 8; nt++)
#pragma unroll
      for (int r = 0; r < 8; r++) oacc[nt][r] *= alpha[r];

    // P = exp(S - m); write bf16 strip to LDS (own wave's rows only)
    float rsum[8];
#pragma unroll
    for (int r = 0; r < 8; r++) rsum[r] = 0.0f;
#pragma unroll
    for (int ct = 0; ct < 4; ct++) {
#pragma unroll
      for (int r = 0; r < 8; r++) {
        float pv = __expf(sacc[ct][r] - mrow[r]);
        rsum[r] += pv;
        Ps[(wid * 16 + r + 8 * h) * 64 + ct * 16 + lm] = (__bf16)pv;
      }
    }
#pragma unroll
    for (int r = 0; r < 8; r++) {
      float s = rsum[r];
      for (int m = 1; m < 16; m <<= 1) s += __shfl_xor(s, m, 32);
      lrow[r] += s;
    }

    // O += P @ V  (A = P from LDS, B = V^T from LDS)
    const __bf16* pr = Ps + (wid * 16 + lm) * 64;
#pragma unroll
    for (int ks = 0; ks < 2; ks++) {
      v8bf plo = *(const v8bf*)(pr + ks * 32 + 8 * h);
      v8bf phi = *(const v8bf*)(pr + ks * 32 + 8 * h + 16);
      v16bf pf = cat16(plo, phi);
      v16bf vfr[8];
#pragma unroll
      for (int nt = 0; nt < 8; nt++) {
        const __bf16* vb = Vt + (nt * 16 + lm) * 64 + ks * 32 + 8 * h;
        v8bf blo = *(const v8bf*)vb;
        v8bf bhi = *(const v8bf*)(vb + 16);
        vfr[nt] = cat16(blo, bhi);
      }
#pragma unroll
      for (int nt = 0; nt < 8; nt++) oacc[nt] = wmma_bf16(pf, vfr[nt], oacc[nt]);
    }
  }

  // epilogue: O / l, write bf16
#pragma unroll
  for (int r = 0; r < 8; r++) {
    int qr = q0 + wid * 16 + r + 8 * h;
    __bf16* op = O + ((size_t)(b * SS + qr) * HQ_ + hq) * HD_;
    float inv = 1.0f / lrow[r];
#pragma unroll
    for (int nt = 0; nt < 8; nt++) op[nt * 16 + lm] = (__bf16)(oacc[nt][r] * inv);
  }
}

// ---------------- router: gate rms -> logits -> softmax -> top2 -> combine + lb terms ----------------
__global__ void router_kernel(const float* __restrict__ hmat, const float* __restrict__ router_scale,
                              const float* __restrict__ w_gate,
                              float* __restrict__ combine, float* __restrict__ lbacc) {
  int row = blockIdx.x;
  const float* hr = hmat + (size_t)row * DD;
  int tid = threadIdx.x;
  float ss = 0.0f;
  for (int c = tid; c < DD; c += 256) { float v = hr[c]; ss += v * v; }
  for (int m = 1; m < 32; m <<= 1) ss += __shfl_xor(ss, m, 32);
  __shared__ float sred[8];
  if ((tid & 31) == 0) sred[tid >> 5] = ss;
  __syncthreads();
  float tot = 0.0f;
#pragma unroll
  for (int i = 0; i < 8; i++) tot += sred[i];
  float rms = rsqrtf(tot / (float)DD + EPSF);
  float dm = rms * 0.022097086912079608f;  // D^-0.5

  float acc[EE];
#pragma unroll
  for (int e = 0; e < EE; e++) acc[e] = 0.0f;
  for (int c = tid; c < DD; c += 256) {
    float g = hr[c] * dm * router_scale[c];
    const float* wg = w_gate + (size_t)c * EE;
#pragma unroll
    for (int e = 0; e < EE; e++) acc[e] += g * wg[e];
  }
  __shared__ float sacc[8][EE];
#pragma unroll
  for (int e = 0; e < EE; e++) {
    float a = acc[e];
    for (int m = 1; m < 32; m <<= 1) a += __shfl_xor(a, m, 32);
    if ((tid & 31) == 0) sacc[tid >> 5][e] = a;
  }
  __syncthreads();
  if (tid == 0) {
    float lg[EE];
#pragma unroll
    for (int e = 0; e < EE; e++) {
      float s = 0.0f;
      for (int w = 0; w < 8; w++) s += sacc[w][e];
      lg[e] = s;
    }
    float mx = lg[0];
    for (int e = 1; e < EE; e++) mx = fmaxf(mx, lg[e]);
    float ex[EE], sum = 0.0f;
    for (int e = 0; e < EE; e++) { ex[e] = __expf(lg[e] - mx); sum += ex[e]; }
    float pr[EE];
    for (int e = 0; e < EE; e++) pr[e] = ex[e] / sum;
    int i1 = 0;
    for (int e = 1; e < EE; e++) if (pr[e] > pr[i1]) i1 = e;
    int i2 = (i1 == 0) ? 1 : 0;
    for (int e = 0; e < EE; e++) if (e != i1 && pr[e] > pr[i2]) i2 = e;
    float w1 = pr[i1], w2 = pr[i2], wn = w1 + w2;
    float* cb = combine + (size_t)row * EE;
    for (int e = 0; e < EE; e++) cb[e] = 0.0f;
    cb[i1] = w1 / wn;
    cb[i2] = w2 / wn;
    atomicAdd(&lbacc[i1], 1.0f);
    atomicAdd(&lbacc[i2], 1.0f);
    for (int e = 0; e < EE; e++) atomicAdd(&lbacc[EE + e], pr[e]);
  }
}

__global__ void expert_combine_kernel(float* __restrict__ racc, const float* __restrict__ eout,
                                      const float* __restrict__ combine, int e) {
  int row = blockIdx.x;
  float w = combine[(size_t)row * EE + e];
  for (int c = threadIdx.x; c < DD; c += 256) {
    size_t i = (size_t)row * DD + c;
    racc[i] += w * eout[i];
  }
}

__global__ void final_kernel(float* __restrict__ out, const float* __restrict__ rn,
                             const float* __restrict__ shn, const float* __restrict__ hmat,
                             const float* __restrict__ ls, long long n) {
  long long i = (long long)blockIdx.x * blockDim.x + threadIdx.x;
  long long stride = (long long)gridDim.x * blockDim.x;
  float s = ls[0];
  for (; i < n; i += stride) out[i] = (rn[i] + shn[i] + hmat[i]) * s;
}

__global__ void lb_final_kernel(float* __restrict__ out, const float* __restrict__ acc) {
  float s = 0.0f;
  for (int e = 0; e < EE; e++) s += (acc[e] / 4096.0f) * (acc[EE + e] / 4096.0f);
  out[0] = s * 8.0f;   // mean over E (=/8) times E^2 (=64)
}

// ---------------- host orchestration ----------------
extern "C" void kernel_launch(void* const* d_in, const int* in_sizes, int n_in,
                              void* d_out, int out_size, void* d_ws, size_t ws_size,
                              hipStream_t stream) {
  (void)in_sizes; (void)n_in; (void)out_size; (void)ws_size;

  const int*   positions      = (const int*)d_in[0];
  const float* x              = (const float*)d_in[1];
  const float* pre_attn_scale = (const float*)d_in[2];
  const float* wq             = (const float*)d_in[3];
  const float* wk             = (const float*)d_in[4];
  const float* wv             = (const float*)d_in[5];
  const float* wo             = (const float*)d_in[6];
  const float* q_norm_scale   = (const float*)d_in[7];
  const float* k_norm_scale   = (const float*)d_in[8];
  const float* v_norm_scale   = (const float*)d_in[9];
  const float* post_attn_scale= (const float*)d_in[10];
  const float* pre_ffw_scale  = (const float*)d_in[11];
  const float* shared_wi0     = (const float*)d_in[12];
  const float* shared_wi1     = (const float*)d_in[13];
  const float* shared_wo      = (const float*)d_in[14];
  const float* post_ff1_scale = (const float*)d_in[15];
  const float* pre_ff2_scale  = (const float*)d_in[16];
  const float* router_scale   = (const float*)d_in[17];
  const float* w_gate         = (const float*)d_in[18];
  const float* ewi0           = (const float*)d_in[19];
  const float* ewi1           = (const float*)d_in[20];
  const float* ewo            = (const float*)d_in[21];
  const float* post_ff2_scale = (const float*)d_in[22];
  const float* layer_scalar   = (const float*)d_in[23];

  float* out = (float*)d_out;
  const long long OUTN = (long long)M4 * DD;   // 8388608

  // deterministic bump allocator on workspace
  char* base = (char*)d_ws;
  size_t off = 0;
  auto alloc = [&](size_t bytes) -> void* {
    off = (off + 255) & ~(size_t)255;
    void* p = base + off;
    off += bytes;
    return p;
  };

  // bf16 weights
  __bf16* wqb   = (__bf16*)alloc((size_t)DD * DD * 2);
  __bf16* wkb   = (__bf16*)alloc((size_t)DD * HKV_ * HD_ * 2);
  __bf16* wvb   = (__bf16*)alloc((size_t)DD * HKV_ * HD_ * 2);
  __bf16* wob   = (__bf16*)alloc((size_t)DD * DD * 2);
  __bf16* swi0b = (__bf16*)alloc((size_t)DD * FS_ * 2);
  __bf16* swi1b = (__bf16*)alloc((size_t)DD * FS_ * 2);
  __bf16* swob  = (__bf16*)alloc((size_t)FS_ * DD * 2);
  __bf16* ewi0b = (__bf16*)alloc((size_t)EE * DD * FE_ * 2);
  __bf16* ewi1b = (__bf16*)alloc((size_t)EE * DD * FE_ * 2);
  __bf16* ewob  = (__bf16*)alloc((size_t)EE * FE_ * DD * 2);
  // bf16 activations
  __bf16* lnxb  = (__bf16*)alloc((size_t)M4 * DD * 2);
  __bf16* qb    = (__bf16*)alloc((size_t)M4 * DD * 2);
  __bf16* kb    = (__bf16*)alloc((size_t)M4 * HKV_ * HD_ * 2);
  __bf16* vb    = (__bf16*)alloc((size_t)M4 * HKV_ * HD_ * 2);
  __bf16* ob    = (__bf16*)alloc((size_t)M4 * DD * 2);
  __bf16* ffinb = (__bf16*)alloc((size_t)M4 * DD * 2);
  __bf16* sgb   = (__bf16*)alloc((size_t)M4 * FS_ * 2);
  __bf16* rinb  = (__bf16*)alloc((size_t)M4 * DD * 2);
  __bf16* abf   = (__bf16*)alloc((size_t)M4 * FE_ * 2);
  // f32 scratch
  float* qf   = (float*)alloc((size_t)M4 * DD * 4);
  float* kf   = (float*)alloc((size_t)M4 * HKV_ * HD_ * 4);
  float* vf   = (float*)alloc((size_t)M4 * HKV_ * HD_ * 4);
  float* attn = (float*)alloc((size_t)M4 * DD * 4);
  float* hbuf = (float*)alloc((size_t)M4 * DD * 4);
  float* g0   = (float*)alloc((size_t)M4 * FS_ * 4);   // reused as expert a0
  float* g1   = (float*)alloc((size_t)M4 * FS_ * 4);   // reused as expert a1
  float* spre = (float*)alloc((size_t)M4 * DD * 4);
  float* shn  = (float*)alloc((size_t)M4 * DD * 4);
  float* eout = (float*)alloc((size_t)M4 * DD * 4);
  float* racc = (float*)alloc((size_t)M4 * DD * 4);
  float* rn   = (float*)alloc((size_t)M4 * DD * 4);
  float* combine = (float*)alloc((size_t)M4 * EE * 4);
  float* lbacc   = (float*)alloc(16 * 4);

  // --- weight conversions to bf16 ---
  cvt_kernel<<<2048, 256, 0, stream>>>(wq,         wqb,   (long long)DD * DD);
  cvt_kernel<<<2048, 256, 0, stream>>>(wk,         wkb,   (long long)DD * HKV_ * HD_);
  cvt_kernel<<<2048, 256, 0, stream>>>(wv,         wvb,   (long long)DD * HKV_ * HD_);
  cvt_kernel<<<2048, 256, 0, stream>>>(wo,         wob,   (long long)DD * DD);
  cvt_kernel<<<4096, 256, 0, stream>>>(shared_wi0, swi0b, (long long)DD * FS_);
  cvt_kernel<<<4096, 256, 0, stream>>>(shared_wi1, swi1b, (long long)DD * FS_);
  cvt_kernel<<<4096, 256, 0, stream>>>(shared_wo,  swob,  (long long)FS_ * DD);
  cvt_kernel<<<4096, 256, 0, stream>>>(ewi0,       ewi0b, (long long)EE * DD * FE_);
  cvt_kernel<<<4096, 256, 0, stream>>>(ewi1,       ewi1b, (long long)EE * DD * FE_);
  cvt_kernel<<<4096, 256, 0, stream>>>(ewo,        ewob,  (long long)EE * FE_ * DD);

  // --- attention block ---
  rmsnorm_kernel<<<M4, 256, 0, stream>>>(x, pre_attn_scale, nullptr, nullptr, lnxb, DD, 1.0f);
  gemm_bf16_kernel<<<dim3(16, 32), 256, 0, stream>>>(lnxb, wqb, qf, M4, DD,  DD);
  gemm_bf16_kernel<<<dim3(8,  32), 256, 0, stream>>>(lnxb, wkb, kf, M4, 1024, DD);
  gemm_bf16_kernel<<<dim3(8,  32), 256, 0, stream>>>(lnxb, wvb, vf, M4, 1024, DD);
  headnorm_rope_kernel<<<M4 * HQ_,  128, 0, stream>>>(qf, q_norm_scale, positions, qb, HQ_,  1);
  headnorm_rope_kernel<<<M4 * HKV_, 128, 0, stream>>>(kf, k_norm_scale, positions, kb, HKV_, 1);
  headnorm_rope_kernel<<<M4 * HKV_, 128, 0, stream>>>(vf, v_norm_scale, positions, vb, HKV_, 0);
  attn_kernel<<<dim3(SS / 128, BB * HQ_), 256, 0, stream>>>(qb, kb, vb, ob);
  gemm_bf16_kernel<<<dim3(16, 32), 256, 0, stream>>>(ob, wob, attn, M4, DD, DD);
  // h = rms(attn)*post_attn_scale + x
  rmsnorm_kernel<<<M4, 256, 0, stream>>>(attn, post_attn_scale, x, hbuf, nullptr, DD, 1.0f);

  // --- shared FFN ---
  rmsnorm_kernel<<<M4, 256, 0, stream>>>(hbuf, pre_ffw_scale, nullptr, nullptr, ffinb, DD, 1.0f);
  gemm_bf16_kernel<<<dim3(64, 32), 256, 0, stream>>>(ffinb, swi0b, g0, M4, FS_, DD);
  gemm_bf16_kernel<<<dim3(64, 32), 256, 0, stream>>>(ffinb, swi1b, g1, M4, FS_, DD);
  geglu_kernel<<<8192, 256, 0, stream>>>(g0, g1, sgb, (long long)M4 * FS_);
  gemm_bf16_kernel<<<dim3(16, 32), 256, 0, stream>>>(sgb, swob, spre, M4, DD, FS_);
  rmsnorm_kernel<<<M4, 256, 0, stream>>>(spre, post_ff1_scale, nullptr, shn, nullptr, DD, 1.0f);

  // --- MoE ---
  rmsnorm_kernel<<<M4, 256, 0, stream>>>(hbuf, pre_ff2_scale, nullptr, nullptr, rinb, DD, 1.0f);
  zero_kernel<<<1, 64, 0, stream>>>(lbacc, 16);
  zero_kernel<<<4096, 256, 0, stream>>>(racc, OUTN);
  router_kernel<<<M4, 256, 0, stream>>>(hbuf, router_scale, w_gate, combine, lbacc);
  for (int e = 0; e < EE; e++) {
    const __bf16* w0 = ewi0b + (size_t)e * DD * FE_;
    const __bf16* w1 = ewi1b + (size_t)e * DD * FE_;
    const __bf16* w2 = ewob  + (size_t)e * FE_ * DD;
    gemm_bf16_kernel<<<dim3(8, 32), 256, 0, stream>>>(rinb, w0, g0, M4, FE_, DD);
    gemm_bf16_kernel<<<dim3(8, 32), 256, 0, stream>>>(rinb, w1, g1, M4, FE_, DD);
    geglu_kernel<<<4096, 256, 0, stream>>>(g0, g1, abf, (long long)M4 * FE_);
    gemm_bf16_kernel<<<dim3(16, 32), 256, 0, stream>>>(abf, w2, eout, M4, DD, FE_);
    expert_combine_kernel<<<M4, 256, 0, stream>>>(racc, eout, combine, e);
  }
  rmsnorm_kernel<<<M4, 256, 0, stream>>>(racc, post_ff2_scale, nullptr, rn, nullptr, DD, 1.0f);

  // --- output ---
  final_kernel<<<8192, 256, 0, stream>>>(out, rn, shn, hbuf, layer_scalar, OUTN);
  lb_final_kernel<<<1, 1, 0, stream>>>(out + OUTN, lbacc);
}
